// GridNetBlock_21268678050128
// MI455X (gfx1250) — matrix operand order repaired and checked by reading, not compiled, not run on hardware
//
#include <hip/hip_runtime.h>
#include <cmath>

// ---------------------------------------------------------------------------
// GridNet block for MI455X (gfx1250, wave32, WMMA).
// GEMMs: f32 operands staged into padded bf16 (zero-filled, pre-transposed)
// buffers; single-layout WMMA kernel runs v_wmma_f32_16x16x32_bf16 with
// unguarded global_load_b128 fragment loads, software-pipelined with 2-deep
// ping-pong buffers (no register rotation copies).
// ---------------------------------------------------------------------------

typedef __attribute__((ext_vector_type(16))) __bf16 v16bf;
typedef __attribute__((ext_vector_type(8)))  __bf16 v8bf;
typedef __attribute__((ext_vector_type(8)))  float  v8f;

#define EPSV 1e-5f

static __device__ inline __bf16 f2bf(float f) {
  unsigned u = __builtin_bit_cast(unsigned, f);
  unsigned r = u + 0x7fffu + ((u >> 16) & 1u);   // round-to-nearest-even
  unsigned short h = (unsigned short)(r >> 16);
  return __builtin_bit_cast(__bf16, h);
}

static __device__ inline float siluf(float x) { return x / (1.f + __expf(-x)); }

// ---------------------------------------------------------------------------
// f32 -> padded bf16 staging.  dst is (Mp x Kp) row-major per batch, zero-pad.
//   transposed==0 : element (m,k) = src[m*lda + k]
//   transposed==1 : element (m,k) = src[k*lda + m]
// ---------------------------------------------------------------------------
__global__ void cvt_bf16_kernel(const float* __restrict__ src, __bf16* __restrict__ dst,
                                int M, int K, int lda, int Mp, int Kp,
                                long long sSrc, int transposed) {
  long long id = (long long)blockIdx.x * blockDim.x + threadIdx.x;
  long long tot = (long long)Mp * Kp;
  if (id >= tot) return;
  int bat = blockIdx.y;
  int m = (int)(id / Kp), k = (int)(id % Kp);
  float v = 0.f;
  if (m < M && k < K) {
    const float* s = src + (long long)bat * sSrc;
    v = transposed ? s[(long long)k * lda + m] : s[(long long)m * lda + k];
  }
  dst[(long long)bat * tot + id] = f2bf(v);
}

// ---------------------------------------------------------------------------
// WMMA GEMM on staged bf16:  C[m,n] = sum_k A[m,k] * B[n,k]   (both row-major,
// K padded to 32, rows padded to 64).  Block = 4 waves, tile 64(M) x 64(N).
// 2-deep ping-pong software pipeline: slab k+32 loads overlap slab k WMMAs.
// ---------------------------------------------------------------------------

#define LOAD_FRAGS(A0, A1, B0, B1, KOFF)                         \
  do {                                                           \
    A0 = *(const v8bf*)(arow + (KOFF));                          \
    A1 = *(const v8bf*)(arow + (KOFF) + 16);                     \
    _Pragma("unroll")                                            \
    for (int t = 0; t < 4; ++t) {                                \
      const __bf16* bp = brow + (long long)t * 16 * Kp + (KOFF); \
      B0[t] = *(const v8bf*)bp;                                  \
      B1[t] = *(const v8bf*)(bp + 16);                           \
    }                                                            \
  } while (0)

#define WMMA_STEP(A0, A1, B0, B1)                                        \
  do {                                                                   \
    v16bf af;                                                            \
    _Pragma("unroll")                                                    \
    for (int j = 0; j < 8; ++j) { af[j] = A0[j]; af[j + 8] = A1[j]; }    \
    _Pragma("unroll")                                                    \
    for (int t = 0; t < 4; ++t) {                                        \
      v16bf bf_;                                                         \
      _Pragma("unroll")                                                  \
      for (int j = 0; j < 8; ++j) { bf_[j] = B0[t][j]; bf_[j + 8] = B1[t][j]; } \
      acc[t] = __builtin_amdgcn_wmma_f32_16x16x32_bf16(                  \
          false, af, false, bf_, (short)0, acc[t], false, false);        \
    }                                                                    \
  } while (0)

__global__ __launch_bounds__(128) void wmma_gemm_bf_kernel(
    const __bf16* __restrict__ Ag, const __bf16* __restrict__ Bg,
    float* __restrict__ Cg, int M, int N, int Kp, int ldc,
    long long psA, long long psB, long long sC) {
  const __bf16* A = Ag + (long long)blockIdx.z * psA;
  const __bf16* B = Bg + (long long)blockIdx.z * psB;
  float*        C = Cg + (long long)blockIdx.z * sC;

  const int wave = threadIdx.x >> 5;
  const int lane = threadIdx.x & 31;
  const int m0 = blockIdx.y * 64 + wave * 16;
  const int n0 = blockIdx.x * 64;
  const int mloc = lane & 15;
  const int hi   = lane >> 4;

  const __bf16* arow = A + (long long)(m0 + mloc) * Kp + hi * 8;
  const __bf16* brow = B + (long long)(n0 + mloc) * Kp + hi * 8;

  v8f acc[4];
#pragma unroll
  for (int t = 0; t < 4; ++t) acc[t] = v8f{};

  v8bf pa0, pa1, pb0[4], pb1[4];   // ping
  v8bf qa0, qa1, qb0[4], qb1[4];   // pong

  LOAD_FRAGS(pa0, pa1, pb0, pb1, 0);

  int k0 = 0;
  for (; k0 + 64 < Kp; k0 += 64) {
    LOAD_FRAGS(qa0, qa1, qb0, qb1, k0 + 32);
    WMMA_STEP(pa0, pa1, pb0, pb1);
    LOAD_FRAGS(pa0, pa1, pb0, pb1, k0 + 64);
    WMMA_STEP(qa0, qa1, qb0, qb1);
  }
  if (k0 + 32 < Kp) {   // two slabs remain: k0, k0+32
    LOAD_FRAGS(qa0, qa1, qb0, qb1, k0 + 32);
    WMMA_STEP(pa0, pa1, pb0, pb1);
    WMMA_STEP(qa0, qa1, qb0, qb1);
  } else {              // one slab remains
    WMMA_STEP(pa0, pa1, pb0, pb1);
  }

  // ---- store: VGPR r -> M = m0 + r + 8*hi ; lane -> N = n0 + 16t + mloc
#pragma unroll
  for (int t = 0; t < 4; ++t) {
#pragma unroll
    for (int r = 0; r < 8; ++r) {
      int mm = m0 + r + (hi << 3);
      int nn = n0 + t * 16 + mloc;
      if (mm < M && nn < N) C[(long long)mm * ldc + nn] = acc[t][r];
    }
  }
}

// ---------------------------------------------------------------------------
// Elementwise / reduction kernels
// ---------------------------------------------------------------------------

__global__ void ln4d_kernel(const float* __restrict__ x, const float* __restrict__ g,
                            const float* __restrict__ b, float* __restrict__ o,
                            int C, int P) {
  int p = blockIdx.x * blockDim.x + threadIdx.x;
  if (p >= P) return;
  float s = 0.f, s2 = 0.f;
  for (int c = 0; c < C; ++c) { float v = x[(long long)c * P + p]; s += v; s2 += v * v; }
  float mu = s / C, var = s2 / C - mu * mu;
  float rinv = rsqrtf(var + EPSV);
  for (int c = 0; c < C; ++c)
    o[(long long)c * P + p] = (x[(long long)c * P + p] - mu) * rinv * g[c] + b[c];
}

__global__ void unfold_intra_kernel(const float* __restrict__ h, float* __restrict__ u,
                                    int Nb, int L) {
  long long id = (long long)blockIdx.x * blockDim.x + threadIdx.x;
  long long tot = (long long)Nb * L * 256;
  if (id >= tot) return;
  int c4 = id % 256; long long r = id / 256;
  int lo = r % L;    int n = r / L;
  int c = c4 >> 2, k = c4 & 3;
  u[id] = h[(long long)c * 8320 + n * 65 + lo + k];
}

__global__ void unfold_inter_kernel(const float* __restrict__ h, float* __restrict__ u,
                                    int Nb, int L) {
  long long id = (long long)blockIdx.x * blockDim.x + threadIdx.x;
  long long tot = (long long)Nb * L * 256;
  if (id >= tot) return;
  int c4 = id % 256; long long r = id / 256;
  int lo = r % L;    int n = r / L;
  int c = c4 >> 2, k = c4 & 3;
  u[id] = h[(long long)c * 8320 + (long long)(lo + k) * 65 + n];
}

__global__ void flip_kernel(const float* __restrict__ u, float* __restrict__ uf,
                            int Nb, int L, int D) {
  long long id = (long long)blockIdx.x * blockDim.x + threadIdx.x;
  long long tot = (long long)Nb * L * D;
  if (id >= tot) return;
  int d = id % D; long long r = id / D;
  int l = r % L; int n = r / L;
  uf[id] = u[((long long)n * L + (L - 1 - l)) * D + d];
}

__global__ void rmsnorm_kernel(const float* __restrict__ u, const float* __restrict__ w,
                               float* __restrict__ o, int M) {
  int warp = threadIdx.x >> 5, lane = threadIdx.x & 31;
  int row = blockIdx.x * (blockDim.x >> 5) + warp;
  if (row >= M) return;
  const float* ur = u + (long long)row * 256;
  float s = 0.f;
  for (int j = lane; j < 256; j += 32) { float v = ur[j]; s += v * v; }
  for (int off = 16; off; off >>= 1) s += __shfl_xor(s, off);
  float r = rsqrtf(s / 256.f + EPSV);
  float* orow = o + (long long)row * 256;
  for (int j = lane; j < 256; j += 32) orow[j] = ur[j] * r * w[j];
}

__global__ void conv_silu_kernel(const float* __restrict__ xz, const float* __restrict__ cw,
                                 const float* __restrict__ cb, float* __restrict__ xs,
                                 int Nb, int L) {
  long long id = (long long)blockIdx.x * blockDim.x + threadIdx.x;
  long long tot = (long long)Nb * L * 1024;
  if (id >= tot) return;
  int d = id % 1024; long long m = id / 1024;
  int l = m % L; int n = m / L;
  float s = cb[d];
#pragma unroll
  for (int k = 0; k < 4; ++k) {
    int ls = l - 3 + k;
    if (ls >= 0) s += cw[d * 4 + k] * xz[((long long)n * L + ls) * 2048 + d];
  }
  xs[id] = siluf(s);
}

__global__ void scan_kernel(const float* __restrict__ dbl, const float* __restrict__ dtw,
                            const float* __restrict__ dtb, const float* __restrict__ alog,
                            const float* __restrict__ Dp, const float* __restrict__ xz,
                            float* __restrict__ xs, int Nb, int L) {
  int id = blockIdx.x * blockDim.x + threadIdx.x;
  if (id >= Nb * 1024) return;
  int d = id % 1024, n = id / 1024;
  float a[16], w[16], h[16];
#pragma unroll
  for (int s = 0; s < 16; ++s) {
    a[s] = -__expf(alog[d * 16 + s]);
    w[s] = dtw[d * 16 + s];
    h[s] = 0.f;
  }
  float dbv = dtb[d], dpv = Dp[d];
  for (int l = 0; l < L; ++l) {
    long long m = (long long)n * L + l;
    const float* db = dbl + m * 48;
    float dtr = dbv;
#pragma unroll
    for (int r = 0; r < 16; ++r) dtr += db[r] * w[r];
    float dt = (dtr > 20.f) ? dtr : log1pf(__expf(dtr));
    float ut = xs[m * 1024 + d];
    float y = 0.f;
#pragma unroll
    for (int s = 0; s < 16; ++s) {
      float dA = __expf(dt * a[s]);
      h[s] = h[s] * dA + (dt * ut) * db[16 + s];
      y += h[s] * db[32 + s];
    }
    y += ut * dpv;
    float z = xz[m * 2048 + 1024 + d];
    y *= siluf(z);
    xs[m * 1024 + d] = y;
  }
}

__global__ void concat_resid_kernel(const float* __restrict__ ymix, const float* __restrict__ U,
                                    float* __restrict__ hcat, int Nb, int L, int off, int rev) {
  long long id = (long long)blockIdx.x * blockDim.x + threadIdx.x;
  long long tot = (long long)Nb * L * 256;
  if (id >= tot) return;
  int c = id % 256; long long m = id / 256;
  int l = m % L; int n = m / L;
  int lp = rev ? (L - 1 - l) : l;
  hcat[((long long)n * L + lp) * 512 + off + c] = ymix[id] + U[id];
}

__global__ void ct_scatter_intra_kernel(const float* __restrict__ Z, const float* __restrict__ b,
                                        const float* __restrict__ resid, float* __restrict__ out,
                                        int L) {
  int id = blockIdx.x * blockDim.x + threadIdx.x;
  if (id >= 64 * 128 * 65) return;
  int f = id % 65; int t = (id / 65) % 128; int o = id / (65 * 128);
  long long pos = (long long)o * 8320 + t * 65 + f;
  float s = b[o] + resid[pos];
#pragma unroll
  for (int k = 0; k < 4; ++k) {
    int l = f - k;
    if (l >= 0 && l < L) s += Z[((long long)t * L + l) * 256 + o * 4 + k];
  }
  out[pos] = s;
}

__global__ void ct_scatter_inter_kernel(const float* __restrict__ Z, const float* __restrict__ b,
                                        const float* __restrict__ resid, float* __restrict__ out,
                                        int L) {
  int id = blockIdx.x * blockDim.x + threadIdx.x;
  if (id >= 64 * 128 * 65) return;
  int f = id % 65; int t = (id / 65) % 128; int o = id / (65 * 128);
  long long pos = (long long)o * 8320 + t * 65 + f;
  float s = b[o] + resid[pos];
#pragma unroll
  for (int k = 0; k < 4; ++k) {
    int l = t - k;
    if (l >= 0 && l < L) s += Z[((long long)f * L + l) * 256 + o * 4 + k];
  }
  out[pos] = s;
}

__global__ void qkv_norm_kernel(const float* __restrict__ y, const float* __restrict__ bias,
                                const float* __restrict__ aalpha, const float* __restrict__ g,
                                const float* __restrict__ bt, float* __restrict__ outn, int E) {
  __shared__ float rA[256], rB[256];
  int h = blockIdx.x / 128, t = blockIdx.x % 128;
  int EF = E * 65, ld = 4 * E;
  float s = 0.f, s2 = 0.f;
  for (int i = threadIdx.x; i < EF; i += 256) {
    int e = i / 65, f = i % 65;
    float v = y[((long long)t * 65 + f) * ld + h * E + e] + bias[h * E + e];
    v = (v >= 0.f) ? v : aalpha[h] * v;
    s += v; s2 += v * v;
  }
  rA[threadIdx.x] = s; rB[threadIdx.x] = s2;
  __syncthreads();
  for (int st = 128; st > 0; st >>= 1) {
    if (threadIdx.x < st) { rA[threadIdx.x] += rA[threadIdx.x + st]; rB[threadIdx.x] += rB[threadIdx.x + st]; }
    __syncthreads();
  }
  float mu = rA[0] / EF;
  float rinv = rsqrtf(rB[0] / EF - mu * mu + EPSV);
  for (int i = threadIdx.x; i < EF; i += 256) {
    int e = i / 65, f = i % 65;
    float v = y[((long long)t * 65 + f) * ld + h * E + e] + bias[h * E + e];
    v = (v >= 0.f) ? v : aalpha[h] * v;
    outn[(long long)h * 128 * EF + (long long)t * EF + e * 65 + f] =
        (v - mu) * rinv * g[(h * E + e) * 65 + f] + bt[(h * E + e) * 65 + f];
  }
}

__global__ void softmax_kernel(float* __restrict__ S, float scale) {
  __shared__ float red[128];
  int row = blockIdx.x, tid = threadIdx.x;
  float v = S[(long long)row * 128 + tid] * scale;
  red[tid] = v; __syncthreads();
  for (int st = 64; st > 0; st >>= 1) {
    if (tid < st) red[tid] = fmaxf(red[tid], red[tid + st]);
    __syncthreads();
  }
  float mx = red[0]; __syncthreads();
  float e = __expf(v - mx);
  red[tid] = e; __syncthreads();
  for (int st = 64; st > 0; st >>= 1) {
    if (tid < st) red[tid] += red[tid + st];
    __syncthreads();
  }
  S[(long long)row * 128 + tid] = e / red[0];
}

__global__ void reorder_o_kernel(const float* __restrict__ O, float* __restrict__ Oc) {
  int id = blockIdx.x * blockDim.x + threadIdx.x;
  if (id >= 532480) return;
  int c = id % 64; int m = id / 64;
  int f = m % 65, t = m / 65;
  int h = c >> 4, dv = c & 15;
  Oc[id] = O[(long long)h * 133120 + (long long)t * 1040 + dv * 65 + f];
}

__global__ void final_norm_kernel(const float* __restrict__ yp, const float* __restrict__ bias,
                                  const float* __restrict__ alpha, const float* __restrict__ g,
                                  const float* __restrict__ bt, const float* __restrict__ resid,
                                  float* __restrict__ out) {
  __shared__ float rA[256], rB[256];
  int t = blockIdx.x;
  const int CF = 64 * 65;
  float a = alpha[0];
  float s = 0.f, s2 = 0.f;
  for (int i = threadIdx.x; i < CF; i += 256) {
    int o = i / 65, f = i % 65;
    float v = yp[((long long)t * 65 + f) * 64 + o] + bias[o];
    v = (v >= 0.f) ? v : a * v;
    s += v; s2 += v * v;
  }
  rA[threadIdx.x] = s; rB[threadIdx.x] = s2;
  __syncthreads();
  for (int st = 128; st > 0; st >>= 1) {
    if (threadIdx.x < st) { rA[threadIdx.x] += rA[threadIdx.x + st]; rB[threadIdx.x] += rB[threadIdx.x + st]; }
    __syncthreads();
  }
  float mu = rA[0] / CF;
  float rinv = rsqrtf(rB[0] / CF - mu * mu + EPSV);
  for (int i = threadIdx.x; i < CF; i += 256) {
    int o = i / 65, f = i % 65;
    float v = yp[((long long)t * 65 + f) * 64 + o] + bias[o];
    v = (v >= 0.f) ? v : a * v;
    long long pos = (long long)o * 8320 + t * 65 + f;
    out[pos] = (v - mu) * rinv * g[o * 65 + f] + bt[o * 65 + f] + resid[pos];
  }
}

// ---------------------------------------------------------------------------
// Host side
// ---------------------------------------------------------------------------

static inline int ceil64(int x) { return (x + 63) & ~63; }
static inline int ceil32(int x) { return (x + 31) & ~31; }

static void gemm_bf(const float* A, const float* B, float* C,
                    int M, int N, int K, int lda, int ldb, int ldc,
                    bool ta, bool tb_nn, int batch,
                    long long sA, long long sB, long long sC,
                    __bf16* Abf, __bf16* Bbf, hipStream_t st) {
  int Mp = ceil64(M), Np = ceil64(N), Kp = ceil32(K);
  {
    long long tot = (long long)Mp * Kp;
    dim3 g((unsigned)((tot + 255) / 256), batch);
    cvt_bf16_kernel<<<g, 256, 0, st>>>(A, Abf, M, K, lda, Mp, Kp, sA, ta ? 1 : 0);
  }
  {
    long long tot = (long long)Np * Kp;
    dim3 g((unsigned)((tot + 255) / 256), batch);
    cvt_bf16_kernel<<<g, 256, 0, st>>>(B, Bbf, N, K, ldb, Np, Kp, sB, tb_nn ? 1 : 0);
  }
  dim3 grid(Np / 64, Mp / 64, batch), blk(128);
  wmma_gemm_bf_kernel<<<grid, blk, 0, st>>>(Abf, Bbf, C, M, N, Kp, ldc,
                                            (long long)Mp * Kp, (long long)Np * Kp, sC);
}

struct MambaP { const float *norm_w, *in_proj, *conv_w, *conv_b, *x_proj, *dt_w, *dt_b, *A_log, *Dp, *out_proj; };

static void run_mamba_dir(const float* U, const MambaP& p, int Nb, int L,
                          float* ur, float* xz, float* xs, float* dbl, float* hcat,
                          __bf16* Abf, __bf16* Bbf, int off, int rev, hipStream_t st) {
  int M = Nb * L;
  rmsnorm_kernel<<<(M + 7) / 8, 256, 0, st>>>(U, p.norm_w, ur, M);
  gemm_bf(ur, p.in_proj, xz, M, 2048, 256, 256, 256, 2048, false, false, 1, 0, 0, 0, Abf, Bbf, st);
  {
    long long tot = (long long)M * 1024;
    conv_silu_kernel<<<(int)((tot + 255) / 256), 256, 0, st>>>(xz, p.conv_w, p.conv_b, xs, Nb, L);
  }
  gemm_bf(xs, p.x_proj, dbl, M, 48, 1024, 1024, 1024, 48, false, false, 1, 0, 0, 0, Abf, Bbf, st);
  scan_kernel<<<(Nb * 1024 + 255) / 256, 256, 0, st>>>(dbl, p.dt_w, p.dt_b, p.A_log, p.Dp, xz, xs, Nb, L);
  gemm_bf(xs, p.out_proj, ur, M, 256, 1024, 1024, 1024, 256, false, false, 1, 0, 0, 0, Abf, Bbf, st);
  {
    long long tot = (long long)M * 256;
    concat_resid_kernel<<<(int)((tot + 255) / 256), 256, 0, st>>>(ur, U, hcat, Nb, L, off, rev);
  }
}

extern "C" void kernel_launch(void* const* d_in, const int* in_sizes, int n_in,
                              void* d_out, int out_size, void* d_ws, size_t ws_size,
                              hipStream_t stream) {
  auto P = [&](int i) { return (const float*)d_in[i]; };
  auto mp = [&](int base) {
    MambaP m{P(base), P(base + 1), P(base + 2), P(base + 3), P(base + 4),
             P(base + 5), P(base + 6), P(base + 7), P(base + 8), P(base + 9)};
    return m;
  };

  const float* x = P(0);
  float* W = (float*)d_ws;
  size_t off = 0;
  auto alloc = [&](size_t n) { float* p = W + off; off += n; return p; };
  float* hln   = alloc(532480);
  float* u     = alloc(2080000);   // 8125*256 max
  float* uflip = alloc(2080000);
  float* ur    = alloc(2080000);   // reused as ymix
  float* xz    = alloc(16640000);  // 8125*2048
  float* xs    = alloc(8320000);   // 8125*1024
  float* dbl   = alloc(390000);    // 8125*48
  float* hcat  = alloc(4160000);   // 8125*512
  float* Z     = alloc(2080000);   // 8125*256
  float* intra = alloc(532480);
  float* inter = alloc(532480);
  float* yq    = alloc(266240);    // 8320*32
  float* yk    = alloc(266240);
  float* yv    = alloc(532480);    // 8320*64
  float* Qn    = alloc(266240);    // 4*128*520
  float* Kn    = alloc(266240);
  float* Vn    = alloc(532480);    // 4*128*1040
  float* Sm    = alloc(65536);     // 4*128*128
  float* Ob    = alloc(532480);
  float* Oc    = alloc(532480);
  float* yproj = alloc(532480);
  __bf16* Abf = (__bf16*)alloc(4161536);  // 8128*1024 bf16 max
  __bf16* Bbf = (__bf16*)alloc(294912);   // 589,824 bf16 max (4x1088x128)
  (void)ws_size; (void)n_in; (void)in_sizes; (void)out_size;

  // ---------------- intra stage (over F, batch = T) ----------------
  {
    int Nb = 128, L = 62, M = Nb * L;
    ln4d_kernel<<<(8320 + 255) / 256, 256, 0, stream>>>(x, P(1), P(2), hln, 64, 8320);
    long long tu = (long long)M * 256;
    unfold_intra_kernel<<<(int)((tu + 255) / 256), 256, 0, stream>>>(hln, u, Nb, L);
    flip_kernel<<<(int)((tu + 255) / 256), 256, 0, stream>>>(u, uflip, Nb, L, 256);
    run_mamba_dir(u,     mp(5),  Nb, L, ur, xz, xs, dbl, hcat, Abf, Bbf, 0,   0, stream);
    run_mamba_dir(uflip, mp(15), Nb, L, ur, xz, xs, dbl, hcat, Abf, Bbf, 256, 1, stream);
    gemm_bf(hcat, P(45), Z, M, 256, 512, 512, 256, 256, false, true, 1, 0, 0, 0, Abf, Bbf, stream);
    ct_scatter_intra_kernel<<<(532480 + 255) / 256, 256, 0, stream>>>(Z, P(46), x, intra, L);
  }

  // ---------------- inter stage (over T, batch = F) ----------------
  {
    int Nb = 65, L = 125, M = Nb * L;
    ln4d_kernel<<<(8320 + 255) / 256, 256, 0, stream>>>(intra, P(3), P(4), hln, 64, 8320);
    long long tu = (long long)M * 256;
    unfold_inter_kernel<<<(int)((tu + 255) / 256), 256, 0, stream>>>(hln, u, Nb, L);
    flip_kernel<<<(int)((tu + 255) / 256), 256, 0, stream>>>(u, uflip, Nb, L, 256);
    run_mamba_dir(u,     mp(25), Nb, L, ur, xz, xs, dbl, hcat, Abf, Bbf, 0,   0, stream);
    run_mamba_dir(uflip, mp(35), Nb, L, ur, xz, xs, dbl, hcat, Abf, Bbf, 256, 1, stream);
    gemm_bf(hcat, P(47), Z, M, 256, 512, 512, 256, 256, false, true, 1, 0, 0, 0, Abf, Bbf, stream);
    ct_scatter_inter_kernel<<<(532480 + 255) / 256, 256, 0, stream>>>(Z, P(48), intra, inter, L);
  }

  // ---------------- attention ----------------
  {
    gemm_bf(inter, P(49), yq, 8320, 32, 64, 8320, 64, 32, true, false, 1, 0, 0, 0, Abf, Bbf, stream);
    gemm_bf(inter, P(54), yk, 8320, 32, 64, 8320, 64, 32, true, false, 1, 0, 0, 0, Abf, Bbf, stream);
    gemm_bf(inter, P(59), yv, 8320, 64, 64, 8320, 64, 64, true, false, 1, 0, 0, 0, Abf, Bbf, stream);
    qkv_norm_kernel<<<4 * 128, 256, 0, stream>>>(yq, P(50), P(51), P(52), P(53), Qn, 8);
    qkv_norm_kernel<<<4 * 128, 256, 0, stream>>>(yk, P(55), P(56), P(57), P(58), Kn, 8);
    qkv_norm_kernel<<<4 * 128, 256, 0, stream>>>(yv, P(60), P(61), P(62), P(63), Vn, 16);

    gemm_bf(Qn, Kn, Sm, 128, 128, 520, 520, 520, 128, false, false,
            4, 66560, 66560, 16384, Abf, Bbf, stream);
    softmax_kernel<<<4 * 128, 128, 0, stream>>>(Sm, 1.0f / sqrtf(520.0f));
    gemm_bf(Sm, Vn, Ob, 128, 1040, 128, 128, 1040, 1040, false, true,
            4, 16384, 133120, 133120, Abf, Bbf, stream);
    reorder_o_kernel<<<(532480 + 255) / 256, 256, 0, stream>>>(Ob, Oc);
    gemm_bf(Oc, P(64), yproj, 8320, 64, 64, 64, 64, 64, false, false, 1, 0, 0, 0, Abf, Bbf, stream);
    final_norm_kernel<<<128, 256, 0, stream>>>(yproj, P(65), P(66), P(67), P(68),
                                               inter, (float*)d_out);
  }
}
